// ResidualVectorQuantizer_19215683682941
// MI455X (gfx1250) — matrix operand order repaired and checked by reading, not compiled, not run on hardware
//
#include <hip/hip_runtime.h>
#include <hip/hip_bf16.h>

// ---------------- problem constants ----------------
#define BB   8
#define DD   256
#define TT   2048
#define NQ   8
#define KK   2048
#define NN   (BB * TT)          // 16384 rows
#define ND   (NN * DD)          // quantized_out elements
#define ROWS 128                // rows per workgroup (8 waves x 16)
#define CTW  64                 // codebook columns per LDS tile
#define BTPAD (DD + 8)          // f16 tile row stride (528 B, bank-conflict pad)

typedef __attribute__((ext_vector_type(16))) _Float16 v16h;
typedef __attribute__((ext_vector_type(8)))  _Float16 v8h;
typedef __attribute__((ext_vector_type(8)))  float    v8f;

// ---------------- prep: zero losses, ||e_k||^2, optional f16 codebook ----------------
__global__ void rvq_prep(const float* __restrict__ embeds,
                         float* __restrict__ ws_loss,
                         float* __restrict__ ws_esq,
                         _Float16* __restrict__ ws_e16) {
    if (blockIdx.x == 0 && threadIdx.x < NQ) ws_loss[threadIdx.x] = 0.0f;
    int e = blockIdx.x * blockDim.x + threadIdx.x;   // 0 .. NQ*KK-1
    if (e < NQ * KK) {
        const float4* p = (const float4*)(embeds + (size_t)e * DD);
        float s = 0.0f;
        if (ws_e16) {
            _Float16* dst = ws_e16 + (size_t)e * DD;
#pragma unroll 8
            for (int i = 0; i < DD / 4; ++i) {
                float4 v = p[i];
                s += v.x * v.x + v.y * v.y + v.z * v.z + v.w * v.w;
                dst[4 * i + 0] = (_Float16)v.x;
                dst[4 * i + 1] = (_Float16)v.y;
                dst[4 * i + 2] = (_Float16)v.z;
                dst[4 * i + 3] = (_Float16)v.w;
            }
        } else {
#pragma unroll 8
            for (int i = 0; i < DD / 4; ++i) {
                float4 v = p[i];
                s += v.x * v.x + v.y * v.y + v.z * v.z + v.w * v.w;
            }
        }
        ws_esq[e] = s;
    }
}

// ---------------- fused RVQ main kernel ----------------
__global__ __launch_bounds__(256) void rvq_main(const float* __restrict__ x,
                                                const float* __restrict__ embeds,
                                                const _Float16* __restrict__ e16,
                                                float* __restrict__ out,
                                                float* __restrict__ ws_loss,
                                                const float* __restrict__ ws_esq) {
    // LDS: f32 residual (persistent across layers) + double-buffered f16 codebook tile
    __shared__ __attribute__((aligned(16))) float    lds_res[ROWS][DD + 4];   // 1040B stride
    __shared__ __attribute__((aligned(16))) _Float16 lds_bt[2][CTW][BTPAD];   // 528B stride

    const int tid     = threadIdx.x;
    const int lane    = tid & 31;
    const int wave    = tid >> 5;
    const int lane16  = lane & 15;
    const int hi      = lane >> 4;            // half-wave (A/B K-range select)
    const int rowBase = blockIdx.x * ROWS;
    const int waveRow = wave * 16;
    const bool useAsync = (e16 != nullptr);

    // ---- load xt (transpose of x[B,D,T]) into residual, coalesced over t ----
    {
        const int b  = rowBase / TT;
        const int t0 = rowBase % TT;
        for (int i = tid; i < ROWS * DD; i += 256) {
            int r = i & (ROWS - 1);
            int d = i >> 7;
            lds_res[r][d] = x[((size_t)b * DD + d) * TT + (t0 + r)];
        }
    }
    __syncthreads();

    v16h afrag[8];   // this wave's 16 rows x 256 depth, f16

    for (int q = 0; q < NQ; ++q) {
        const float* eq  = embeds + (size_t)q * KK * DD;
        const float* esq = ws_esq + q * KK;

        // ---- build A fragments (reads only this wave's own rows) ----
        {
            const float* rowp = lds_res[waveRow + lane16];
#pragma unroll
            for (int f = 0; f < 8; ++f) {
                const int kb = f * 32;
                const float* p0 = rowp + kb + hi * 8;
                const float* p1 = rowp + kb + 16 + hi * 8;
#pragma unroll
                for (int j = 0; j < 8; ++j) {
                    afrag[f][j]     = (_Float16)p0[j];
                    afrag[f][8 + j] = (_Float16)p1[j];
                }
            }
        }

        float best[8];
        int   bidx[8];
#pragma unroll
        for (int j = 0; j < 8; ++j) { best[j] = -3.4e38f; bidx[j] = 0; }

        // score_tile: WMMA on one 16-col subtile group; acc seeded with -0.5*||e||^2
        auto compute_tile = [&](int buf, int colBase) {
#pragma unroll
            for (int c = 0; c < CTW / 16; ++c) {
                const int   col  = colBase + c * 16 + lane16;
                const float init = -0.5f * esq[col];       // L2-resident
                v8f acc = {init, init, init, init, init, init, init, init};
                const _Float16* bp = &lds_bt[buf][c * 16 + lane16][hi * 8];
#pragma unroll
                for (int f = 0; f < 8; ++f) {
                    v8h b0 = *(const v8h*)(bp + f * 32);
                    v8h b1 = *(const v8h*)(bp + f * 32 + 16);
                    v16h bfrag = __builtin_shufflevector(
                        b0, b1, 0, 1, 2, 3, 4, 5, 6, 7, 8, 9, 10, 11, 12, 13, 14, 15);
                    acc = __builtin_amdgcn_wmma_f32_16x16x32_f16(
                        false, afrag[f], false, bfrag, (short)0, acc, false, false);
                }
#pragma unroll
                for (int j = 0; j < 8; ++j) {              // acc = r.e - 0.5||e||^2
                    if (acc[j] > best[j]) { best[j] = acc[j]; bidx[j] = col; }
                }
            }
        };

        if (useAsync) {
            // ===== async DMA path: f16 codebook streamed via GLOBAL_LOAD_ASYNC_TO_LDS =====
            const _Float16* eq16 = e16 + (size_t)q * KK * DD;
            // 16B chunks: chunk = (i*8 + wave)*32 + lane, i = 0..7 ; row = chunk>>5, c16 = chunk&31
            const unsigned chunk0 = (unsigned)(wave * 32 + lane);
            const unsigned row0   = chunk0 >> 5;
            const unsigned c16    = chunk0 & 31;

            auto issue_tile = [&](int ct, int buf) {
                unsigned g = ((unsigned)(ct * CTW + row0) * DD + c16 * 8) * 2;   // byte off vs eq16
                unsigned l = (unsigned)(uintptr_t)&lds_bt[buf][row0][c16 * 8];   // LDS byte addr
#pragma unroll
                for (int i = 0; i < 8; ++i) {
                    asm volatile("global_load_async_to_lds_b128 %0, %1, %2"
                                 :: "v"(l), "v"(g), "s"(eq16) : "memory");
                    g += 8u * DD * 2;        // 8 rows * 512 B in global
                    l += 8u * BTPAD * 2;     // 8 rows * 528 B in LDS
                }
            };

            issue_tile(0, 0);
            for (int ct = 0; ct < KK / CTW; ++ct) {
                asm volatile("s_wait_asynccnt 0x0" ::: "memory");   // my copies landed
                __syncthreads();                                    // everyone's landed
                if (ct + 1 < KK / CTW) issue_tile(ct + 1, (ct + 1) & 1);
                compute_tile(ct & 1, ct * CTW);                     // overlap with next DMA
            }
        } else {
            // ===== fallback: load f32 codebook, convert, ds_store =====
            for (int ct = 0; ct < KK / CTW; ++ct) {
                const int colBase = ct * CTW;
                __syncthreads();
                for (int i = tid; i < CTW * DD; i += 256) {
                    int rr = i >> 8;
                    int dd = i & (DD - 1);
                    lds_bt[ct & 1][rr][dd] = (_Float16)eq[(size_t)(colBase + rr) * DD + dd];
                }
                if (tid < CTW && ct + 1 < KK / CTW)
                    __builtin_prefetch(eq + (size_t)(colBase + CTW + tid) * DD, 0, 3);
                __syncthreads();
                compute_tile(ct & 1, colBase);
            }
        }

        // ---- argmax reduce across the 16 lanes holding each row ----
#pragma unroll
        for (int m = 1; m < 16; m <<= 1) {
#pragma unroll
            for (int j = 0; j < 8; ++j) {
                float ov = __shfl_xor(best[j], m, 16);
                int   oi = __shfl_xor(bidx[j], m, 16);
                if (ov > best[j] || (ov == best[j] && oi < bidx[j])) {
                    best[j] = ov; bidx[j] = oi;
                }
            }
        }

        // ---- write indices (row m = j + 8*hi) ----
        if (lane16 == 0) {
#pragma unroll
            for (int j = 0; j < 8; ++j) {
                int rg = rowBase + waveRow + hi * 8 + j;
                out[(size_t)ND + (size_t)q * NN + rg] = (float)bidx[j];
            }
        }

        // ---- residual update (exact f32 gather from global codebook) + loss ----
        float lsum = 0.0f;
#pragma unroll
        for (int mrow = 0; mrow < 16; ++mrow) {
            const int j   = mrow & 7;
            const int src = (mrow >> 3) * 16;
            const int idx = __shfl(bidx[j], src, 32);
            const float* ep = eq + (size_t)idx * DD;
            float* rp = lds_res[waveRow + mrow];
#pragma unroll
            for (int it = 0; it < DD / 32; ++it) {
                int   d  = lane + it * 32;
                float rn = rp[d] - ep[d];
                rp[d] = rn;
                lsum += rn * rn;
            }
        }
#pragma unroll
        for (int m = 1; m < 32; m <<= 1) lsum += __shfl_xor(lsum, m, 32);
        if (lane == 0) atomicAdd(&ws_loss[q], lsum);
    }

    __syncthreads();
    // ---- quantized_out = xt - residual_final (transform in LDS, write coalesced) ----
    {
        const int b  = rowBase / TT;
        const int t0 = rowBase % TT;
        for (int i = tid; i < ROWS * DD; i += 256) {
            int r = i & (ROWS - 1);
            int d = i >> 7;
            float xv = x[((size_t)b * DD + d) * TT + (t0 + r)];
            lds_res[r][d] = xv - lds_res[r][d];
        }
        __syncthreads();
        for (int i = tid; i < ROWS * DD; i += 256) {
            int d = i & (DD - 1);
            int r = i >> 8;
            out[((size_t)(rowBase + r)) * DD + d] = lds_res[r][d];
        }
    }
}

// ---------------- finalize: losses = sum / (N*D) ----------------
__global__ void rvq_finalize(const float* __restrict__ ws_loss, float* __restrict__ out) {
    int q = threadIdx.x;
    if (q < NQ)
        out[(size_t)ND + (size_t)NQ * NN + q] = ws_loss[q] * (1.0f / ((float)NN * (float)DD));
}

extern "C" void kernel_launch(void* const* d_in, const int* in_sizes, int n_in,
                              void* d_out, int out_size, void* d_ws, size_t ws_size,
                              hipStream_t stream) {
    (void)in_sizes; (void)n_in; (void)out_size;
    const float* x      = (const float*)d_in[0];
    const float* embeds = (const float*)d_in[1];
    float* out     = (float*)d_out;
    float* ws_loss = (float*)d_ws;          // 8 floats
    float* ws_esq  = (float*)d_ws + 16;     // NQ*KK floats

    const size_t e16_off  = ((16 + NQ * KK) * sizeof(float) + 127) & ~(size_t)127;
    const size_t e16_need = e16_off + (size_t)NQ * KK * DD * sizeof(_Float16);
    _Float16* ws_e16 = (ws_size >= e16_need) ? (_Float16*)((char*)d_ws + e16_off) : nullptr;

    rvq_prep<<<(NQ * KK) / 256, 256, 0, stream>>>(embeds, ws_loss, ws_esq, ws_e16);
    rvq_main<<<NN / ROWS, 256, 0, stream>>>(x, embeds, ws_e16, out, ws_loss, ws_esq);
    rvq_finalize<<<1, 32, 0, stream>>>(ws_loss, out);
}
